// ConvHydra_34677565948587
// MI455X (gfx1250) — matrix operand order repaired and checked by reading, not compiled, not run on hardware
//
#include <hip/hip_runtime.h>
#include <hip/hip_bf16.h>

typedef __bf16 bf16;
typedef __attribute__((ext_vector_type(16))) __bf16 v16bf;
typedef __attribute__((ext_vector_type(8)))  __bf16 v8bf;
typedef __attribute__((ext_vector_type(8)))  float  v8f;

// Model constants
static constexpr int kSeq  = 2048;
static constexpr int kDM   = 128;   // D_MODEL
static constexpr int kNH   = 4;     // heads
static constexpr int kDI   = 256;   // D_INNER
static constexpr int kDS   = 128;   // D_STATE
static constexpr int kDTR  = 8;     // DT_RANK
static constexpr int kBT   = kNH * kSeq;        // 8192 (batch*seq rows)
static constexpr int kDBC  = kDTR + 2 * kDS;    // 264
static constexpr int kImgStride = 2176;         // 2048 + 127, rounded to keep 16B rows

__device__ __forceinline__ float silu_f(float x) { return x / (1.f + __expf(-x)); }

// ---------------------------------------------------------------------------
// Tiled WMMA GEMM with split-K: Cpart(z) = A(MxK,bf16) * B(KxN,bf16) over
// K-range [z*Ksplit, min(K,(z+1)*Ksplit)).  grid.z = #splits, Cpart stride M*N.
// IM2COL: B[k][n] = img[(k>>7)*ldimg + n + (k&127)]  (padded conv image)
// Block 256 threads = 8 waves; tile 128x64; K step 32.
// A tile is streamed with GLOBAL_LOAD_ASYNC_TO_LDS_B128 (ASYNCcnt path).
// ---------------------------------------------------------------------------
template <bool IM2COL>
__global__ void gemm_bf16_kernel(const bf16* __restrict__ A,
                                 const bf16* __restrict__ B,
                                 float* __restrict__ C,
                                 int M, int N, int K, int ldimg, int Ksplit) {
  __shared__ __align__(64) bf16 As[128][32];
  __shared__ __align__(64) bf16 Bs[64][32];   // transposed: [n][k]

  const int tid  = threadIdx.x;
  const int lane = tid & 31;
  const int wave = tid >> 5;
  const int wm   = wave >> 1;      // 0..3
  const int wn   = wave & 1;       // 0..1
  const int lrow = lane & 15;
  const int half = lane >> 4;
  const int m0   = blockIdx.y * 128;
  const int n0   = blockIdx.x * 64;

  const int kStart = blockIdx.z * Ksplit;
  int kEnd = kStart + Ksplit;
  if (kEnd > K) kEnd = K;
  float* Cb = C + (size_t)blockIdx.z * M * N;

  v8f acc[2][2] = {};

  const int ar   = tid >> 1;            // A tile row 0..127
  const int aseg = (tid & 1) << 4;      // 0 or 16 (K segment)
  const int bk   = tid >> 3;            // B tile k 0..31
  const int bc0  = (tid & 7) << 3;      // B tile col base (8 cols)

  const unsigned ldsA = (unsigned)(size_t)&As[ar][aseg];  // low 32 bits = LDS offset
  const int grA = m0 + ar;

  for (int k0 = kStart; k0 < kEnd; k0 += 32) {
    __syncthreads();
    { // ---- A tile: 128x32; async global->LDS copy (2 x b128 per thread)
      if (grA < M) {
        const bf16* ga = A + (size_t)grA * K + k0 + aseg;
#if defined(__AMDGCN__)
        // INST_OFFSET applies to both LDS and global addresses (ISA 08 §4.4)
        asm volatile(
            "global_load_async_to_lds_b128 %0, %1, off\n\t"
            "global_load_async_to_lds_b128 %0, %1, off offset:16"
            :: "v"(ldsA), "v"(ga) : "memory");
#else
        const v8bf* src = (const v8bf*)ga;
        *((v8bf*)&As[ar][aseg])     = src[0];
        *((v8bf*)&As[ar][aseg + 8]) = src[1];
#endif
      } else {
        v8bf z = {};
        *((v8bf*)&As[ar][aseg])     = z;
        *((v8bf*)&As[ar][aseg + 8]) = z;
      }
    }
    if (IM2COL) { // ---- B tile from padded conv image (implicit im2col)
      int kidx = k0 + bk;
      int ci   = kidx >> 7;
      int kk   = kidx & 127;
      const bf16* src = B + (size_t)ci * ldimg + n0 + bc0 + kk;
      __builtin_prefetch(src + 32, 0, 1);   // next K-tile row -> global_prefetch_b8
#pragma unroll
      for (int e = 0; e < 8; ++e) Bs[bc0 + e][bk] = src[e];
    } else {      // ---- plain B tile (row-major KxN), stored transposed
      const bf16* src = B + (size_t)(k0 + bk) * N + n0 + bc0;
      if (n0 + bc0 + 8 <= N) {
        v8bf v = *(const v8bf*)src;
#pragma unroll
        for (int e = 0; e < 8; ++e) Bs[bc0 + e][bk] = v[e];
      } else {
#pragma unroll
        for (int e = 0; e < 8; ++e) {
          int c = n0 + bc0 + e;
          Bs[bc0 + e][bk] = (c < N) ? src[e] : (bf16)0.f;
        }
      }
    }
#if defined(__AMDGCN__)
    asm volatile("s_wait_asynccnt 0" ::: "memory");  // our async A copies landed
#endif
    __syncthreads();

    // ---- fragments (ISA layouts) + 4 WMMAs per wave
    v16bf afrag[2], bfrag[2];
#pragma unroll
    for (int mi = 0; mi < 2; ++mi) {
      const v8bf* ra = (const v8bf*)&As[wm * 32 + mi * 16 + lrow][0];
      // A: elem e -> K = (e/8)*16 + half*8 + (e&7)
      afrag[mi] = __builtin_shufflevector(ra[half], ra[2 + half],
                                          0,1,2,3,4,5,6,7,8,9,10,11,12,13,14,15);
    }
#pragma unroll
    for (int ni = 0; ni < 2; ++ni) {
      // B: elem e -> K = (e/8)*16 + half*8 + (e&7), lane holds column lrow
      const v8bf* rb = (const v8bf*)&Bs[wn * 32 + ni * 16 + lrow][0];
      bfrag[ni] = __builtin_shufflevector(rb[half], rb[2 + half],
                                          0,1,2,3,4,5,6,7,8,9,10,11,12,13,14,15);
    }
#pragma unroll
    for (int mi = 0; mi < 2; ++mi)
#pragma unroll
      for (int ni = 0; ni < 2; ++ni)
        acc[mi][ni] = __builtin_amdgcn_wmma_f32_16x16x32_bf16(
            false, afrag[mi], false, bfrag[ni], (short)0, acc[mi][ni], false, false);
  }

  // ---- store: D elem v -> M = half*8 + v, N = lrow
#pragma unroll
  for (int mi = 0; mi < 2; ++mi)
#pragma unroll
    for (int ni = 0; ni < 2; ++ni) {
      int col   = n0 + wn * 32 + ni * 16 + lrow;
      int rbase = m0 + wm * 32 + mi * 16 + half * 8;
      if (col < N) {
#pragma unroll
        for (int v = 0; v < 8; ++v) {
          int r = rbase + v;
          if (r < M) Cb[(size_t)r * N + col] = acc[mi][ni][v];
        }
      }
    }
}

// Deterministic split-K reduction: dst[i] = sum_s parts[s*n + i]
__global__ void reduce_sum(const float* __restrict__ parts, float* __restrict__ dst,
                           int n, int S) {
  int i = blockIdx.x * 256 + threadIdx.x;
  if (i < n) {
    float a = 0.f;
    for (int s = 0; s < S; ++s) a += parts[(size_t)s * n + i];
    dst[i] = a;
  }
}

// ---------------------------------------------------------------------------
// Elementwise helpers
// ---------------------------------------------------------------------------
__global__ void cvt_f32_bf16(const float* __restrict__ s, bf16* __restrict__ d, int n) {
  int i = blockIdx.x * 256 + threadIdx.x;
  if (i < n) d[i] = (bf16)s[i];
}

// src (R x Cc) f32 -> dst (Cc x R) bf16   (dst[c*R + r] = src[r*Cc + c])
__global__ void cvt_transpose(const float* __restrict__ s, bf16* __restrict__ d, int R, int Cc) {
  int i = blockIdx.x * 256 + threadIdx.x;
  if (i < R * Cc) {
    int r = i / Cc, c = i % Cc;
    d[(size_t)c * R + r] = (bf16)s[i];
  }
}

__global__ void fill_zero_bf16(bf16* __restrict__ d, int n) {
  int i = blockIdx.x * 256 + threadIdx.x;
  if (i < n) d[i] = (bf16)0.f;
}

__global__ void neg_exp_kernel(const float* __restrict__ s, float* __restrict__ d, int n) {
  int i = blockIdx.x * 256 + threadIdx.x;
  if (i < n) d[i] = -__expf(s[i]);
}

// x flat (128 x 2048 reinterpret) -> bf16 padded image (63 left / 64 right zeros)
__global__ void pad_x_kernel(const float* __restrict__ x, bf16* __restrict__ xpad) {
  int i = blockIdx.x * 256 + threadIdx.x;
  if (i < kDM * kSeq) {
    int ci = i >> 11, s = i & 2047;
    xpad[(size_t)ci * kImgStride + 63 + s] = (bf16)x[i];
  }
}

// xp = silu(conv_out + bias[co]), stored as bf16 at the same flat index
__global__ void bias_silu_cvt(const float* __restrict__ in, const float* __restrict__ bias,
                              bf16* __restrict__ out, int n) {
  int i = blockIdx.x * 256 + threadIdx.x;
  if (i < n) {
    int co  = i >> 11;  // cols = 2048
    float v = in[i] + bias[co];
    out[i]  = (bf16)silu_f(v);
  }
}

// Depthwise causal conv (K=7) over xs = xz[:, :256]; reverse=1 -> anti-causal
__global__ void dwconv_silu(const float* __restrict__ xz, const float* __restrict__ w,
                            const float* __restrict__ bconv, float* __restrict__ xc32,
                            bf16* __restrict__ xc16, int reverse) {
  int i = blockIdx.x * 256 + threadIdx.x;
  if (i >= kBT * kDI) return;
  int d  = i & 255;
  int bt = i >> 8;
  int b  = bt >> 11;
  int t  = bt & 2047;
  float acc = bconv[d];
#pragma unroll
  for (int j = 0; j < 7; ++j) {
    int tt = reverse ? (t + 6 - j) : (t - 6 + j);
    if (tt >= 0 && tt < kSeq)
      acc += w[d * 7 + j] * xz[((size_t)b * kSeq + tt) * 512 + d];
  }
  float s = silu_f(acc);
  xc32[i] = s;
  xc16[i] = (bf16)s;
}

// delta = softplus(dbc[:, :8] @ dt_w.T + dt_b)
__global__ void delta_kernel(const float* __restrict__ dbc, const float* __restrict__ dt_w,
                             const float* __restrict__ dt_b, float* __restrict__ dlt) {
  int i = blockIdx.x * 256 + threadIdx.x;
  if (i >= kBT * kDI) return;
  int d  = i & 255;
  int bt = i >> 8;
  const float* r = &dbc[(size_t)bt * kDBC];
  float acc = dt_b[d];
#pragma unroll
  for (int j = 0; j < kDTR; ++j) acc += r[j] * dt_w[d * kDTR + j];
  dlt[i] = (acc > 20.f) ? acc : log1pf(__expf(acc));
}

// Selective scan: one wave per (b,d), 4 states per lane, shuffle-reduce y.
// Writes y_combined = (scan_y + u*Dp[d]) * silu(z) as bf16 for the out_proj GEMM.
__global__ void scan_kernel(const float* __restrict__ dlt, const float* __restrict__ xc,
                            const float* __restrict__ dbc, const float* __restrict__ xz,
                            const float* __restrict__ Aneg, const float* __restrict__ Dp,
                            bf16* __restrict__ yout, int reverse) {
  int wave = (blockIdx.x * blockDim.x + threadIdx.x) >> 5;  // 0..1023
  int lane = threadIdx.x & 31;
  int b = wave >> 8;    // 0..3
  int d = wave & 255;   // 0..255
  float4 a4 = *reinterpret_cast<const float4*>(&Aneg[(size_t)d * kDS + lane * 4]);
  float Dd = Dp[d];
  float h0 = 0.f, h1 = 0.f, h2 = 0.f, h3 = 0.f;
  for (int i = 0; i < kSeq; ++i) {
    int t = reverse ? (kSeq - 1 - i) : i;
    size_t bt = (size_t)b * kSeq + t;
    float dtv = dlt[bt * 256 + d];
    float u   = xc[bt * 256 + d];
    float4 B4 = *reinterpret_cast<const float4*>(&dbc[bt * kDBC + kDTR + lane * 4]);
    float4 C4 = *reinterpret_cast<const float4*>(&dbc[bt * kDBC + kDTR + kDS + lane * 4]);
    float du = dtv * u;
    h0 = __expf(dtv * a4.x) * h0 + du * B4.x;
    h1 = __expf(dtv * a4.y) * h1 + du * B4.y;
    h2 = __expf(dtv * a4.z) * h2 + du * B4.z;
    h3 = __expf(dtv * a4.w) * h3 + du * B4.w;
    float pp = h0 * C4.x + h1 * C4.y + h2 * C4.z + h3 * C4.w;
#pragma unroll
    for (int off = 16; off > 0; off >>= 1) pp += __shfl_xor(pp, off, 32);
    if (lane == 0) {
      float z = xz[bt * 512 + 256 + d];
      yout[bt * 256 + d] = (bf16)((pp + u * Dd) * silu_f(z));
    }
  }
}

// xn = rmsnorm(mf+mb)*w, written into padded bf16 image for conv2 (flat reinterpret)
__global__ void rmsnorm_pad_kernel(const float* __restrict__ mf, const float* __restrict__ mb,
                                   const float* __restrict__ nw, bf16* __restrict__ xnpad) {
  int bt = blockIdx.x;     // 0..8191
  int d  = threadIdx.x;    // 0..127
  float v = mf[(size_t)bt * kDM + d] + mb[(size_t)bt * kDM + d];
  float s = v * v;
#pragma unroll
  for (int off = 16; off > 0; off >>= 1) s += __shfl_xor(s, off, 32);
  __shared__ float red[4];
  if ((d & 31) == 0) red[d >> 5] = s;
  __syncthreads();
  float tot = red[0] + red[1] + red[2] + red[3];
  float xn  = v * rsqrtf(tot * (1.f / 128.f) + 1e-8f) * nw[d];
  size_t f = (size_t)bt * kDM + d;     // flat index == (512 x 2048) reinterpret
  int ci = (int)(f >> 11);
  int sp = (int)(f & 2047);
  xnpad[(size_t)ci * kImgStride + 63 + sp] = (bf16)xn;
}

__global__ void final_silu(const float* __restrict__ in, const float* __restrict__ bias,
                           float* __restrict__ out) {
  int i = blockIdx.x * 256 + threadIdx.x;
  if (i < kDM * kSeq) {
    int co  = i >> 11;
    float v = in[i] + bias[co];
    out[i]  = silu_f(v);
  }
}

// ---------------------------------------------------------------------------
extern "C" void kernel_launch(void* const* d_in, const int* in_sizes, int n_in,
                              void* d_out, int out_size, void* d_ws, size_t ws_size,
                              hipStream_t stream) {
  (void)in_sizes; (void)n_in; (void)out_size; (void)ws_size;
  const float* x        = (const float*)d_in[0];
  const float* cw1      = (const float*)d_in[1];
  const float* cb1      = (const float*)d_in[2];
  const float* cw2      = (const float*)d_in[3];
  const float* cb2      = (const float*)d_in[4];
  const float* in_proj  = (const float*)d_in[5];
  const float* conv_w   = (const float*)d_in[6];
  const float* conv_b   = (const float*)d_in[7];
  const float* x_proj   = (const float*)d_in[8];
  const float* dt_w     = (const float*)d_in[9];
  const float* dt_b     = (const float*)d_in[10];
  const float* A_log    = (const float*)d_in[11];
  const float* Dp       = (const float*)d_in[12];
  const float* out_proj = (const float*)d_in[13];
  const float* norm_w   = (const float*)d_in[14];
  float* out = (float*)d_out;

  char* p = (char*)d_ws;
  auto alloc = [&](size_t bytes) -> char* {
    char* r = p;
    p += (bytes + 255) & ~(size_t)255;
    return r;
  };

  bf16* W1b   = (bf16*)alloc((size_t)512 * 16384 * 2);
  bf16* W2b   = (bf16*)alloc((size_t)128 * 65536 * 2);
  bf16* B2b   = (bf16*)alloc((size_t)128 * 512 * 2);   // in_proj^T
  bf16* B3b   = (bf16*)alloc((size_t)256 * 264 * 2);   // x_proj^T
  bf16* B5b   = (bf16*)alloc((size_t)256 * 128 * 2);   // out_proj^T
  bf16* xpad  = (bf16*)alloc((size_t)128 * kImgStride * 2);
  bf16* xnpad = (bf16*)alloc((size_t)512 * kImgStride * 2);
  bf16* xpb   = (bf16*)alloc((size_t)kBT * 128 * 2);
  float* Aneg = (float*)alloc((size_t)256 * 128 * 4);
  float* out1 = (float*)alloc((size_t)512 * 2048 * 4);
  float* xz   = (float*)alloc((size_t)kBT * 512 * 4);
  // split-K partial buffer, reused by both convs (8*1M == 32*256K floats)
  float* splitb = (float*)alloc((size_t)8 * 1048576 * 4);
  float* xc32[2]; bf16* xc16[2]; float* dbc[2]; float* dlt[2]; bf16* yc[2]; float* mo[2];
  for (int d2 = 0; d2 < 2; ++d2) {
    xc32[d2] = (float*)alloc((size_t)kBT * 256 * 4);
    xc16[d2] = (bf16*)alloc((size_t)kBT * 256 * 2);
    dbc[d2]  = (float*)alloc((size_t)kBT * 264 * 4);
    dlt[d2]  = (float*)alloc((size_t)kBT * 256 * 4);
    yc[d2]   = (bf16*)alloc((size_t)kBT * 256 * 2);
    mo[d2]   = (float*)alloc((size_t)kBT * 128 * 4);
  }
  float* out2 = (float*)alloc((size_t)128 * 2048 * 4);

  dim3 blk(256);
  auto g1 = [](int n) { return dim3((unsigned)((n + 255) / 256)); };

  // Weight prep
  cvt_f32_bf16<<<g1(512 * 16384), blk, 0, stream>>>(cw1, W1b, 512 * 16384);
  cvt_f32_bf16<<<g1(128 * 65536), blk, 0, stream>>>(cw2, W2b, 128 * 65536);
  cvt_transpose<<<g1(512 * 128), blk, 0, stream>>>(in_proj, B2b, 512, 128);
  cvt_transpose<<<g1(264 * 256), blk, 0, stream>>>(x_proj, B3b, 264, 256);
  cvt_transpose<<<g1(128 * 256), blk, 0, stream>>>(out_proj, B5b, 128, 256);
  neg_exp_kernel<<<g1(256 * 128), blk, 0, stream>>>(A_log, Aneg, 256 * 128);
  fill_zero_bf16<<<g1(128 * kImgStride), blk, 0, stream>>>(xpad, 128 * kImgStride);
  fill_zero_bf16<<<g1(512 * kImgStride), blk, 0, stream>>>(xnpad, 512 * kImgStride);
  pad_x_kernel<<<g1(128 * 2048), blk, 0, stream>>>(x, xpad);

  // conv_in as implicit-im2col GEMM: (512 x 16384) x (16384 x 2048), split-K 8
  gemm_bf16_kernel<true><<<dim3(32, 4, 8), blk, 0, stream>>>(
      W1b, xpad, splitb, 512, 2048, 16384, kImgStride, 2048);
  reduce_sum<<<g1(512 * 2048), blk, 0, stream>>>(splitb, out1, 512 * 2048, 8);
  bias_silu_cvt<<<g1(512 * 2048), blk, 0, stream>>>(out1, cb1, xpb, 512 * 2048);

  // xz = xp @ in_proj.T : (8192 x 128) x (128 x 512)  — shared by fwd/bwd
  gemm_bf16_kernel<false><<<dim3(512 / 64, kBT / 128, 1), blk, 0, stream>>>(
      xpb, B2b, xz, kBT, 512, 128, 0, 128);

  for (int dir = 0; dir < 2; ++dir) {
    dwconv_silu<<<g1(kBT * 256), blk, 0, stream>>>(xz, conv_w, conv_b, xc32[dir], xc16[dir], dir);
    // dbc = xc @ x_proj.T : (8192 x 256) x (256 x 264)
    gemm_bf16_kernel<false><<<dim3((264 + 63) / 64, kBT / 128, 1), blk, 0, stream>>>(
        xc16[dir], B3b, dbc[dir], kBT, 264, 256, 0, 256);
    delta_kernel<<<g1(kBT * 256), blk, 0, stream>>>(dbc[dir], dt_w, dt_b, dlt[dir]);
    scan_kernel<<<dim3(128), blk, 0, stream>>>(dlt[dir], xc32[dir], dbc[dir], xz,
                                               Aneg, Dp, yc[dir], dir);
    // out_dir = y @ out_proj.T : (8192 x 256) x (256 x 128)
    gemm_bf16_kernel<false><<<dim3(128 / 64, kBT / 128, 1), blk, 0, stream>>>(
        yc[dir], B5b, mo[dir], kBT, 128, 256, 0, 256);
  }

  // rmsnorm(fwd+bwd) -> padded bf16 image for conv_out
  rmsnorm_pad_kernel<<<dim3(kBT), dim3(128), 0, stream>>>(mo[0], mo[1], norm_w, xnpad);

  // conv_out as implicit-im2col GEMM: (128 x 65536) x (65536 x 2048), split-K 32
  gemm_bf16_kernel<true><<<dim3(32, 1, 32), blk, 0, stream>>>(
      W2b, xnpad, splitb, 128, 2048, 65536, kImgStride, 2048);
  reduce_sum<<<g1(128 * 2048), blk, 0, stream>>>(splitb, out2, 128 * 2048, 32);
  final_silu<<<g1(128 * 2048), blk, 0, stream>>>(out2, cb2, out);
}
